// VisionTransformerRetrieval_64201171140747
// MI455X (gfx1250) — compile-verified, matrix operand order checked by reading, not run on hardware
//
#include <hip/hip_runtime.h>

// ============================================================================
// VisionTransformerRetrieval for MI455X (gfx1250, wave32, WMMA).
// All matmuls via V_WMMA_F32_16X16X32_BF16 (fp32 ref -> bf16 in, fp32 acc).
// GEMM: 256 thr = 8 waves, block tile 64x128, K-step 32, each wave 32x32
// (4 wmma / K-step). Interior tiles: global_load_b128 -> bf16 pack ->
// ds_store_b128 (no exec-mask churn); edge tiles take a guarded scalar path.
// Scores use ld=200 so attention GEMMs stay on the vector path.
// ============================================================================

#define EMBED   768
#define HEADS   12
#define HDIM    64
#define NTOK    196
#define LTOK    197
#define SLD     200      // padded score row stride (divisible by 4)
#define BATCH   8
#define DEPTH   12
#define MLPD    3072
#define DSTATE  16
#define DTRANK  48
#define KCONV   4

typedef __attribute__((ext_vector_type(16))) __bf16 v16bf;
typedef __attribute__((ext_vector_type(8)))  float  v8f;

#define ACT_NONE     0
#define ACT_GELU     1
#define ACT_SOFTPLUS 2
#define ACT_TANH     3

// ---------------------------------------------------------------------------
// helpers
// ---------------------------------------------------------------------------
__device__ inline unsigned pack_bf16(float lo, float hi) {
  unsigned ul = __builtin_bit_cast(unsigned, lo);
  unsigned uh = __builtin_bit_cast(unsigned, hi);
  ul += 0x7FFFu + ((ul >> 16) & 1u);            // RNE
  uh += 0x7FFFu + ((uh >> 16) & 1u);
  return (ul >> 16) | (uh & 0xFFFF0000u);
}

__device__ inline float act_apply(float x, int act) {
  if (act == ACT_GELU) {
    float x3 = x * x * x;
    return 0.5f * x * (1.0f + tanhf(0.7978845608f * (x + 0.044715f * x3)));
  } else if (act == ACT_SOFTPLUS) {
    return (x > 20.0f) ? x : log1pf(__expf(x));
  } else if (act == ACT_TANH) {
    return tanhf(x);
  }
  return x;
}

// ---------------------------------------------------------------------------
// GEMM tile geometry
// ---------------------------------------------------------------------------
#define BM 64
#define BN 128
#define BK 32
#define LDP 20   // uint pairs per LDS row: 16 data + 4 pad (80B/row, 16B-aligned)

// Load a 16(row) x 32(K) bf16 fragment from an LDS tile of packed bf16 pairs.
// ISA 7.12.2: lane&15 = row, lanes>=16 get K offset +8 (pair offset +4);
// VGPR j holds K pairs {2j,2j+1} (j<4, pair j) and {2j+8,2j+9} (j>=4, pair j+4).
__device__ inline v16bf frag_ld(const unsigned* base, int lane) {
  int r  = lane & 15;
  int kp = (lane >> 4) << 2;
  const unsigned* p = base + r * LDP + kp;
  union { v16bf v; unsigned u[8]; } f;
#pragma unroll
  for (int j = 0; j < 8; ++j) f.u[j] = p[(j < 4) ? j : (j + 4)];
  return f.v;
}

// ---------------------------------------------------------------------------
// Batched GEMM:  C[z] = act( A[z] (MxK) * op(W[z]) + bias ) (+ resid)
//   wTrans=1: W is (N,K) row-major  (A @ W^T, the x@W.T pattern)
//   wTrans=0: W is (K,N) row-major  (A @ W,   the attn@V pattern)
// z = zb*H + zh; per-component strides for A/W/C (resid shares C strides).
// ---------------------------------------------------------------------------
__global__ __launch_bounds__(256)
void k_gemm(const float* __restrict__ A, const float* __restrict__ W,
            const float* __restrict__ bias, const float* __restrict__ resid,
            float* __restrict__ C,
            int M, int N, int K, int lda, int ldw, int ldc,
            int H,
            long long sAb, long long sAh, long long sWb, long long sWh,
            long long sCb, long long sCh,
            int wTrans, int act)
{
  __shared__ unsigned As[BM * LDP];   //  5120 B
  __shared__ unsigned Bs[BN * LDP];   // 10240 B

  int z  = blockIdx.z;
  int zb = z / H, zh = z % H;
  const float* Ab = A + zb * sAb + zh * sAh;
  const float* Wb = W + zb * sWb + zh * sWh;
  float*       Cb = C + zb * sCb + zh * sCh;
  const float* Rb = resid ? (resid + zb * sCb + zh * sCh) : (const float*)0;

  int tid  = threadIdx.x;
  int lane = tid & 31;
  int wid  = tid >> 5;
  int wm   = wid & 1;        // 2 wave rows  (32 M each)
  int wn   = wid >> 1;       // 4 wave cols  (32 N each)
  int m0   = blockIdx.y * BM;
  int n0   = blockIdx.x * BN;

  // staging coordinates (uniform per thread)
  int ar  = tid >> 2;             // A row 0..63
  int apb = (tid & 3) * 4;        // A pair base 0/4/8/12
  int bn  = tid >> 1;             // B row 0..127
  int bpb = (tid & 1) * 8;        // B pair base 0/8

  bool aRowFull = (m0 + BM <= M) && ((lda & 3) == 0);
  bool bRowFull = wTrans && (n0 + BN <= N) && ((ldw & 3) == 0);

  v8f acc00 = {}, acc01 = {}, acc10 = {}, acc11 = {};

  for (int k0 = 0; k0 < K; k0 += BK) {
    bool kFull = (k0 + BK <= K);
    // ---------------- stage A tile (64 x 32) ----------------
    if (aRowFull && kFull) {
      const float* src = Ab + (long long)(m0 + ar) * lda + k0 + apb * 2;
      float4 f0 = *(const float4*)(src);
      float4 f1 = *(const float4*)(src + 4);
      if (k0 + BK < K) __builtin_prefetch(src + BK, 0, 1);
      uint4 o;
      o.x = pack_bf16(f0.x, f0.y);
      o.y = pack_bf16(f0.z, f0.w);
      o.z = pack_bf16(f1.x, f1.y);
      o.w = pack_bf16(f1.z, f1.w);
      *(uint4*)&As[ar * LDP + apb] = o;
    } else {
      int gm = m0 + ar;
      uint4 o;
      unsigned* po = (unsigned*)&o;
#pragma unroll
      for (int j = 0; j < 4; ++j) {
        int gk = k0 + (apb + j) * 2;
        float lo = (gm < M && gk     < K) ? Ab[(long long)gm * lda + gk]     : 0.0f;
        float hi = (gm < M && gk + 1 < K) ? Ab[(long long)gm * lda + gk + 1] : 0.0f;
        po[j] = pack_bf16(lo, hi);
      }
      *(uint4*)&As[ar * LDP + apb] = o;
    }
    // ---------------- stage B tile as Bs[n][k] (128 x 32) ----------------
    if (bRowFull && kFull) {
      const float* src = Wb + (long long)(n0 + bn) * ldw + k0 + bpb * 2;
      float4 f0 = *(const float4*)(src);
      float4 f1 = *(const float4*)(src + 4);
      float4 f2 = *(const float4*)(src + 8);
      float4 f3 = *(const float4*)(src + 12);
      if (k0 + BK < K) __builtin_prefetch(src + BK, 0, 1);
      uint4 o0, o1;
      o0.x = pack_bf16(f0.x, f0.y); o0.y = pack_bf16(f0.z, f0.w);
      o0.z = pack_bf16(f1.x, f1.y); o0.w = pack_bf16(f1.z, f1.w);
      o1.x = pack_bf16(f2.x, f2.y); o1.y = pack_bf16(f2.z, f2.w);
      o1.z = pack_bf16(f3.x, f3.y); o1.w = pack_bf16(f3.z, f3.w);
      *(uint4*)&Bs[bn * LDP + bpb]     = o0;
      *(uint4*)&Bs[bn * LDP + bpb + 4] = o1;
    } else {
      int gn = n0 + bn;
      uint4 o0, o1;
      unsigned* po = (unsigned*)&o0;
#pragma unroll
      for (int j = 0; j < 8; ++j) {
        if (j == 4) po = (unsigned*)&o1;
        int gk = k0 + (bpb + j) * 2;
        float lo = 0.0f, hi = 0.0f;
        if (gn < N) {
          if (gk < K)
            lo = wTrans ? Wb[(long long)gn * ldw + gk]
                        : Wb[(long long)gk * ldw + gn];
          if (gk + 1 < K)
            hi = wTrans ? Wb[(long long)gn * ldw + gk + 1]
                        : Wb[(long long)(gk + 1) * ldw + gn];
        }
        po[j & 3] = pack_bf16(lo, hi);
      }
      *(uint4*)&Bs[bn * LDP + bpb]     = o0;
      *(uint4*)&Bs[bn * LDP + bpb + 4] = o1;
    }
    __syncthreads();

    v16bf a0 = frag_ld(&As[(wm * 32)      * LDP], lane);
    v16bf a1 = frag_ld(&As[(wm * 32 + 16) * LDP], lane);
    v16bf b0 = frag_ld(&Bs[(wn * 32)      * LDP], lane);
    v16bf b1 = frag_ld(&Bs[(wn * 32 + 16) * LDP], lane);
    acc00 = __builtin_amdgcn_wmma_f32_16x16x32_bf16(false, a0, false, b0, (short)0, acc00, false, false);
    acc01 = __builtin_amdgcn_wmma_f32_16x16x32_bf16(false, a0, false, b1, (short)0, acc01, false, false);
    acc10 = __builtin_amdgcn_wmma_f32_16x16x32_bf16(false, a1, false, b0, (short)0, acc10, false, false);
    acc11 = __builtin_amdgcn_wmma_f32_16x16x32_bf16(false, a1, false, b1, (short)0, acc11, false, false);
    __syncthreads();
  }

  // ---- epilogue. C/D layout: VGPR v -> M = v (+8 for lanes>=16), N = lane&15
  int nl   = lane & 15;
  int moff = (lane >> 4) * 8;
  const v8f* accs[4] = { &acc00, &acc01, &acc10, &acc11 };
#pragma unroll
  for (int t = 0; t < 4; ++t) {
    int mbase = m0 + wm * 32 + (t >> 1) * 16 + moff;
    int n     = n0 + wn * 32 + (t & 1) * 16 + nl;
    if (n >= N) continue;
    float bn_ = bias ? bias[n] : 0.0f;
#pragma unroll
    for (int v = 0; v < 8; ++v) {
      int m = mbase + v;
      if (m >= M) continue;
      float val = (*accs[t])[v] + bn_;
      val = act_apply(val, act);
      if (Rb) val += Rb[(long long)m * ldc + n];
      Cb[(long long)m * ldc + n] = val;
    }
  }
}

// ---------------------------------------------------------------------------
// small kernels
// ---------------------------------------------------------------------------
__global__ __launch_bounds__(256)
void k_patch(const float* __restrict__ x, float* __restrict__ xp, int total) {
  int i = blockIdx.x * 256 + threadIdx.x;
  if (i >= total) return;
  int d = i % 768;
  int n = (i / 768) % NTOK;
  int b = i / (768 * NTOK);
  int c  = d / 256;
  int py = (d % 256) / 16;
  int px = d % 16;
  int gy = n / 14, gx = n % 14;
  xp[i] = x[(((long long)(b * 3 + c) * 224) + gy * 16 + py) * 224 + gx * 16 + px];
}

__global__ __launch_bounds__(256)
void k_build_h(const float* __restrict__ tok, const float* __restrict__ cls,
               const float* __restrict__ pos, float* __restrict__ h, int total) {
  int i = blockIdx.x * 256 + threadIdx.x;
  if (i >= total) return;
  int d = i % EMBED;
  int r = (i / EMBED) % LTOK;
  int b = i / (EMBED * LTOK);
  float v = (r == 0) ? cls[d] : tok[((long long)(b * NTOK) + (r - 1)) * EMBED + d];
  h[i] = v + pos[(long long)r * EMBED + d];
}

__global__ __launch_bounds__(256)
void k_mix_ai(const float* __restrict__ h, const float* __restrict__ prev,
              const float* __restrict__ alpha, float* __restrict__ ai, int total) {
  int i = blockIdx.x * 256 + threadIdx.x;
  if (i >= total) return;
  int d = i % EMBED;
  int l = (i / EMBED) % NTOK;
  int b = i / (EMBED * NTOK);
  float t = h[((long long)(b * LTOK + 1 + l)) * EMBED + d];
  if (prev) {
    float al = alpha[0];
    ai[i] = al * prev[i] + (1.0f - al) * t;
  } else {
    ai[i] = t;
  }
}

__global__ __launch_bounds__(256)
void k_conv_silu(const float* __restrict__ xz, const float* __restrict__ cw,
                 const float* __restrict__ cb, float* __restrict__ us, int total) {
  int i = blockIdx.x * 256 + threadIdx.x;
  if (i >= total) return;
  int d = i % EMBED;
  int l = (i / EMBED) % NTOK;
  int b = i / (EMBED * NTOK);
  float s = cb[d];
#pragma unroll
  for (int k = 0; k < KCONV; ++k) {
    int lk = l - (KCONV - 1) + k;
    if (lk >= 0)
      s += xz[((long long)(b * NTOK + lk)) * (2 * EMBED) + d] * cw[d * KCONV + k];
  }
  us[i] = s / (1.0f + __expf(-s));
}

__global__ __launch_bounds__(256)
void k_expA(const float* __restrict__ alog, float* __restrict__ an, int total) {
  int i = blockIdx.x * 256 + threadIdx.x;
  if (i < total) an[i] = -__expf(alog[i]);
}

// selective scan fused with + D*u and * silu(z). one thread per (b, d).
__global__ __launch_bounds__(256)
void k_scan(const float* __restrict__ dtb, const float* __restrict__ us,
            const float* __restrict__ dbl, const float* __restrict__ xz,
            const float* __restrict__ An, const float* __restrict__ Dv,
            float* __restrict__ y) {
  int i = blockIdx.x * 256 + threadIdx.x;
  if (i >= BATCH * EMBED) return;
  int d = i % EMBED;
  int b = i / EMBED;
  float hst[DSTATE];
#pragma unroll
  for (int s = 0; s < DSTATE; ++s) hst[s] = 0.0f;
  const float* Ad = An + d * DSTATE;
  float Dd = Dv[d];
  for (int t = 0; t < NTOK; ++t) {
    long long base = (long long)b * NTOK + t;
    float dt = dtb[base * EMBED + d];
    float u  = us[base * EMBED + d];
    const float* Bc = dbl + base * 80 + DTRANK;
    const float* Cc = dbl + base * 80 + DTRANK + DSTATE;
    float acc = 0.0f;
#pragma unroll
    for (int s = 0; s < DSTATE; ++s) {
      hst[s] = hst[s] * __expf(dt * Ad[s]) + dt * u * Bc[s];
      acc += hst[s] * Cc[s];
    }
    float zv = xz[base * (2 * EMBED) + EMBED + d];
    float sz = zv / (1.0f + __expf(-zv));
    y[base * EMBED + d] = (acc + Dd * u) * sz;
  }
}

__global__ __launch_bounds__(256)
void k_zero(float* p, int n) {
  int i = blockIdx.x * 256 + threadIdx.x;
  if (i < n) p[i] = 0.0f;
}

__global__ __launch_bounds__(256)
void k_route_dot(const float* __restrict__ prevQ, const float* __restrict__ cK,
                 float* __restrict__ scores, int nElem) {
  int j = blockIdx.y;
  const float* q = prevQ + (long long)j * nElem;
  float s = 0.0f;
  for (int i = blockIdx.x * 256 + threadIdx.x; i < nElem; i += gridDim.x * 256)
    s += q[i] * cK[i];
  __shared__ float red[256];
  red[threadIdx.x] = s; __syncthreads();
  for (int t = 128; t > 0; t >>= 1) {
    if (threadIdx.x < t) red[threadIdx.x] += red[threadIdx.x + t];
    __syncthreads();
  }
  if (threadIdx.x == 0) atomicAdd(&scores[j], red[0]);
}

__global__ void k_topk(const float* __restrict__ scores, int n, int k,
                       int* __restrict__ sel) {
  if (threadIdx.x != 0 || blockIdx.x != 0) return;
  float s[DEPTH]; bool used[DEPTH];
  for (int i = 0; i < n; ++i) { s[i] = scores[i]; used[i] = false; }
  for (int kk = 0; kk < k; ++kk) {
    float best = -3.4e38f; int bi = 0;
    for (int i = 0; i < n; ++i)
      if (!used[i] && s[i] > best) { best = s[i]; bi = i; }
    used[bi] = true;
    sel[kk] = bi;
  }
}

__global__ __launch_bounds__(256)
void k_fuse(const float* __restrict__ qcur, const float* __restrict__ prevQ,
            const int* __restrict__ sel, int k, float* __restrict__ out, int nElem) {
  int i = blockIdx.x * 256 + threadIdx.x;
  if (i >= nElem) return;
  float v = qcur[i];
  for (int kk = 0; kk < k; ++kk)
    v += prevQ[(long long)sel[kk] * nElem + i];
  out[i] = v / (1.0f + (float)k);
}

__global__ __launch_bounds__(256)
void k_softmax(float* __restrict__ S, int cols, int ldr, float scale) {
  long long row = blockIdx.x;
  float* p = S + row * ldr;
  __shared__ float red[256];
  float mx = -3.4e38f;
  for (int c = threadIdx.x; c < cols; c += 256) mx = fmaxf(mx, p[c] * scale);
  red[threadIdx.x] = mx; __syncthreads();
  for (int t = 128; t > 0; t >>= 1) {
    if (threadIdx.x < t) red[threadIdx.x] = fmaxf(red[threadIdx.x], red[threadIdx.x + t]);
    __syncthreads();
  }
  float m = red[0]; __syncthreads();
  float sum = 0.0f;
  for (int c = threadIdx.x; c < cols; c += 256) {
    float e = __expf(p[c] * scale - m);
    p[c] = e; sum += e;
  }
  red[threadIdx.x] = sum; __syncthreads();
  for (int t = 128; t > 0; t >>= 1) {
    if (threadIdx.x < t) red[threadIdx.x] += red[threadIdx.x + t];
    __syncthreads();
  }
  float inv = 1.0f / red[0];
  for (int c = threadIdx.x; c < cols; c += 256) p[c] *= inv;
}

__global__ __launch_bounds__(256)
void k_ln(const float* __restrict__ X, long long xs,
          const float* __restrict__ g, const float* __restrict__ b, float eps,
          float* __restrict__ Y, long long ys, int D) {
  long long row = blockIdx.x;
  const float* xr = X + row * xs;
  float* yr = Y + row * ys;
  __shared__ float r1[256], r2[256];
  float s = 0.0f, s2 = 0.0f;
  for (int i = threadIdx.x; i < D; i += 256) { float v = xr[i]; s += v; s2 += v * v; }
  r1[threadIdx.x] = s; r2[threadIdx.x] = s2; __syncthreads();
  for (int t = 128; t > 0; t >>= 1) {
    if (threadIdx.x < t) { r1[threadIdx.x] += r1[threadIdx.x + t]; r2[threadIdx.x] += r2[threadIdx.x + t]; }
    __syncthreads();
  }
  float mean = r1[0] / D;
  float var  = r2[0] / D - mean * mean;
  float inv  = rsqrtf(var + eps);
  for (int i = threadIdx.x; i < D; i += 256)
    yr[i] = (xr[i] - mean) * inv * g[i] + b[i];
}

// h[:,1+l,:] += LN(av * aw)   (tokens already live in h[:,1:,:])
__global__ __launch_bounds__(256)
void k_avln(const float* __restrict__ av, const float* __restrict__ aw,
            const float* __restrict__ g, const float* __restrict__ b,
            float* __restrict__ h) {
  int row = blockIdx.x;                 // 0 .. B*NTOK-1
  int bi = row / NTOK, l = row % NTOK;
  const float* xr = av + (long long)row * EMBED;
  float* hr = h + ((long long)(bi * LTOK + 1 + l)) * EMBED;
  __shared__ float r1[256], r2[256];
  float s = 0.0f, s2 = 0.0f;
  for (int i = threadIdx.x; i < EMBED; i += 256) {
    float v = xr[i] * aw[i]; s += v; s2 += v * v;
  }
  r1[threadIdx.x] = s; r2[threadIdx.x] = s2; __syncthreads();
  for (int t = 128; t > 0; t >>= 1) {
    if (threadIdx.x < t) { r1[threadIdx.x] += r1[threadIdx.x + t]; r2[threadIdx.x] += r2[threadIdx.x + t]; }
    __syncthreads();
  }
  float mean = r1[0] / EMBED;
  float var  = r2[0] / EMBED - mean * mean;
  float inv  = rsqrtf(var + 1e-5f);
  for (int i = threadIdx.x; i < EMBED; i += 256) {
    float v = xr[i] * aw[i];
    hr[i] += (v - mean) * inv * g[i] + b[i];
  }
}

__global__ __launch_bounds__(256)
void k_bnscale(float* __restrict__ f, const float* __restrict__ g,
               const float* __restrict__ b, int total) {
  int i = blockIdx.x * 256 + threadIdx.x;
  if (i >= total) return;
  int n = i % 256;
  f[i] = f[i] * rsqrtf(1.0f + 1e-5f) * g[n] + b[n];
}

// ---------------------------------------------------------------------------
// host orchestration
// ---------------------------------------------------------------------------
static void gemm(hipStream_t st, const float* A, const float* W, const float* bias,
                 const float* resid, float* C, int M, int N, int K,
                 int lda, int ldw, int ldc, int Z, int H,
                 long long sAb, long long sAh, long long sWb, long long sWh,
                 long long sCb, long long sCh, int wT, int act) {
  dim3 grid((N + BN - 1) / BN, (M + BM - 1) / BM, Z);
  k_gemm<<<grid, dim3(256), 0, st>>>(A, W, bias, resid, C, M, N, K, lda, ldw, ldc,
                                     H, sAb, sAh, sWb, sWh, sCb, sCh, wT, act);
}

static inline dim3 g1(long long n) { return dim3((unsigned)((n + 255) / 256)); }

extern "C" void kernel_launch(void* const* d_in, const int* in_sizes, int n_in,
                              void* d_out, int out_size, void* d_ws, size_t ws_size,
                              hipStream_t stream) {
  (void)in_sizes; (void)n_in; (void)out_size; (void)ws_size;
  auto F = [&](int i) { return (const float*)d_in[i]; };

  const float* x_in   = F(0);
  const float* patchW = F(1);  const float* patchB = F(2);
  const float* pos    = F(3);  const float* cls    = F(4);
  const float* alpha  = F(5);  const float* adW    = F(6);
  const float* convQ  = F(7);  const float* convK  = F(8);  const float* convV = F(9);
  const float* normG  = F(10); const float* normB  = F(11);
  const float* fcW    = F(12); const float* fcB    = F(13);
  const float* bnG    = F(14); const float* bnB    = F(15);
  const float* WG     = F(16);
  const float* a_inW  = F(17); const float* a_cw   = F(18); const float* a_cb  = F(19);
  const float* a_xpW  = F(20); const float* a_dtW  = F(21); const float* a_dtB = F(22);
  const float* a_Alog = F(23); const float* a_D    = F(24); const float* a_outW = F(25);
  auto BLK = [&](int i, int j) { return (const float*)d_in[26 + 12 * i + j]; };
  // j: 0 n1_g 1 n1_b 2 qkv_w 3 qkv_b 4 proj_w 5 proj_b 6 n2_g 7 n2_b
  //    8 fc1_w 9 fc1_b 10 fc2_w 11 fc2_b

  const long long TE = (long long)BATCH * NTOK * EMBED;   // 1,204,224
  const long long HE = (long long)BATCH * LTOK * EMBED;   // 1,210,368

  float* wsp = (float*)d_ws;
  size_t off = 0;
  auto alloc = [&](long long n) { float* p = wsp + off; off += (size_t)((n + 15) & ~15LL); return p; };

  float* xp    = alloc(TE);
  float* tok   = alloc(TE);
  float* h     = alloc(HE);
  float* hn    = alloc(HE);
  float* ai    = alloc(TE);
  float* xz    = alloc((long long)BATCH * NTOK * 2 * EMBED);
  float* usilu = alloc(TE);
  float* dbl   = alloc((long long)BATCH * NTOK * 80);
  float* dtb   = alloc(TE);
  float* ybuf  = alloc(TE);
  float* ao    = alloc(TE);
  float* prevQ = alloc(DEPTH * TE);
  float* cKb   = alloc(TE);
  float* fq    = alloc(TE);
  float* Kb    = alloc(TE);
  float* Vb    = alloc(TE);
  float* sc    = alloc((long long)BATCH * HEADS * LTOK * SLD);
  float* av    = alloc(TE);
  float* obuf  = alloc(HE);
  float* qkv   = alloc((long long)BATCH * LTOK * 3 * EMBED);
  float* mlp   = alloc((long long)BATCH * LTOK * MLPD);
  float* An    = alloc(EMBED * DSTATE);
  float* clsln = alloc(BATCH * EMBED);
  float* fbuf  = alloc(BATCH * 256);
  float* scR   = alloc(16);
  int*   sel   = (int*)alloc(16);

  // ---- patch embed + token sequence ----
  k_patch<<<g1(TE), 256, 0, stream>>>(x_in, xp, (int)TE);
  gemm(stream, xp, patchW, patchB, nullptr, tok, BATCH * NTOK, EMBED, EMBED,
       EMBED, EMBED, EMBED, 1, 1, 0, 0, 0, 0, 0, 0, 1, ACT_NONE);
  k_build_h<<<g1(HE), 256, 0, stream>>>(tok, cls, pos, h, (int)HE);
  k_expA<<<g1(EMBED * DSTATE), 256, 0, stream>>>(a_Alog, An, EMBED * DSTATE);

  for (int i = 0; i < DEPTH; ++i) {
    // -------- Mamba adapter --------
    k_mix_ai<<<g1(TE), 256, 0, stream>>>(h, (i == 0) ? nullptr : ao, alpha, ai, (int)TE);
    gemm(stream, ai, a_inW, nullptr, nullptr, xz, BATCH * NTOK, 2 * EMBED, EMBED,
         EMBED, EMBED, 2 * EMBED, 1, 1, 0, 0, 0, 0, 0, 0, 1, ACT_NONE);
    k_conv_silu<<<g1(TE), 256, 0, stream>>>(xz, a_cw, a_cb, usilu, (int)TE);
    gemm(stream, usilu, a_xpW, nullptr, nullptr, dbl, BATCH * NTOK, 80, EMBED,
         EMBED, EMBED, 80, 1, 1, 0, 0, 0, 0, 0, 0, 1, ACT_NONE);
    gemm(stream, dbl, a_dtW, a_dtB, nullptr, dtb, BATCH * NTOK, EMBED, DTRANK,
         80, DTRANK, EMBED, 1, 1, 0, 0, 0, 0, 0, 0, 1, ACT_SOFTPLUS);
    k_scan<<<g1(BATCH * EMBED), 256, 0, stream>>>(dtb, usilu, dbl, xz, An, a_D, ybuf);
    gemm(stream, ybuf, a_outW, nullptr, nullptr, ao, BATCH * NTOK, EMBED, EMBED,
         EMBED, EMBED, EMBED, 1, 1, 0, 0, 0, 0, 0, 0, 1, ACT_NONE);

    // -------- q_cur / routing --------
    float* qcur = prevQ + (long long)i * TE;
    gemm(stream, ao, convQ, nullptr, nullptr, qcur, BATCH * NTOK, EMBED, EMBED,
         EMBED, EMBED, EMBED, 1, 1, 0, 0, 0, 0, 0, 0, 1, ACT_NONE);
    gemm(stream, ao, convK, nullptr, nullptr, cKb, BATCH * NTOK, EMBED, EMBED,
         EMBED, EMBED, EMBED, 1, 1, 0, 0, 0, 0, 0, 0, 1, ACT_NONE);
    const float* fqp = qcur;
    if (i > 0) {
      k_zero<<<1, 256, 0, stream>>>(scR, 16);
      k_route_dot<<<dim3(64, i), 256, 0, stream>>>(prevQ, cKb, scR, (int)TE);
      int kk = (i < 7) ? i : 7;
      k_topk<<<1, 1, 0, stream>>>(scR, i, kk, sel);
      k_fuse<<<g1(TE), 256, 0, stream>>>(qcur, prevQ, sel, kk, fq, (int)TE);
      fqp = fq;
    }
    gemm(stream, fqp, convK, nullptr, nullptr, Kb, BATCH * NTOK, EMBED, EMBED,
         EMBED, EMBED, EMBED, 1, 1, 0, 0, 0, 0, 0, 0, 1, ACT_NONE);
    gemm(stream, fqp, convV, nullptr, nullptr, Vb, BATCH * NTOK, EMBED, EMBED,
         EMBED, EMBED, EMBED, 1, 1, 0, 0, 0, 0, 0, 0, 1, ACT_NONE);

    // -------- inner attention (196 tokens, 96 (b,h) batches) --------
    gemm(stream, qcur, Kb, nullptr, nullptr, sc, NTOK, NTOK, HDIM,
         EMBED, EMBED, SLD, BATCH * HEADS, HEADS,
         (long long)NTOK * EMBED, HDIM, (long long)NTOK * EMBED, HDIM,
         (long long)HEADS * NTOK * SLD, (long long)NTOK * SLD, 1, ACT_NONE);
    k_softmax<<<dim3(BATCH * HEADS * NTOK), 256, 0, stream>>>(sc, NTOK, SLD, 0.125f);
    gemm(stream, sc, Vb, nullptr, nullptr, av, NTOK, HDIM, NTOK,
         SLD, EMBED, EMBED, BATCH * HEADS, HEADS,
         (long long)HEADS * NTOK * SLD, (long long)NTOK * SLD,
         (long long)NTOK * EMBED, HDIM,
         (long long)NTOK * EMBED, HDIM, 0, ACT_NONE);
    k_avln<<<dim3(BATCH * NTOK), 256, 0, stream>>>(av, adW, normG, normB, h);

    // -------- ViT block --------
    k_ln<<<dim3(BATCH * LTOK), 256, 0, stream>>>(h, EMBED, BLK(i, 0), BLK(i, 1),
                                                 1e-6f, hn, EMBED, EMBED);
    gemm(stream, hn, BLK(i, 2), BLK(i, 3), nullptr, qkv, BATCH * LTOK, 3 * EMBED,
         EMBED, EMBED, EMBED, 3 * EMBED, 1, 1, 0, 0, 0, 0, 0, 0, 1, ACT_NONE);
    gemm(stream, qkv, qkv + EMBED, nullptr, nullptr, sc, LTOK, LTOK, HDIM,
         3 * EMBED, 3 * EMBED, SLD, BATCH * HEADS, HEADS,
         (long long)LTOK * 3 * EMBED, HDIM, (long long)LTOK * 3 * EMBED, HDIM,
         (long long)HEADS * LTOK * SLD, (long long)LTOK * SLD, 1, ACT_NONE);
    k_softmax<<<dim3(BATCH * HEADS * LTOK), 256, 0, stream>>>(sc, LTOK, SLD, 0.125f);
    gemm(stream, sc, qkv + 2 * EMBED, nullptr, nullptr, obuf, LTOK, HDIM, LTOK,
         SLD, 3 * EMBED, EMBED, BATCH * HEADS, HEADS,
         (long long)HEADS * LTOK * SLD, (long long)LTOK * SLD,
         (long long)LTOK * 3 * EMBED, HDIM,
         (long long)LTOK * EMBED, HDIM, 0, ACT_NONE);
    gemm(stream, obuf, BLK(i, 4), BLK(i, 5), h, h, BATCH * LTOK, EMBED, EMBED,
         EMBED, EMBED, EMBED, 1, 1, 0, 0, 0, 0, 0, 0, 1, ACT_NONE);
    k_ln<<<dim3(BATCH * LTOK), 256, 0, stream>>>(h, EMBED, BLK(i, 6), BLK(i, 7),
                                                 1e-6f, hn, EMBED, EMBED);
    gemm(stream, hn, BLK(i, 8), BLK(i, 9), nullptr, mlp, BATCH * LTOK, MLPD, EMBED,
         EMBED, EMBED, MLPD, 1, 1, 0, 0, 0, 0, 0, 0, 1, ACT_GELU);
    gemm(stream, mlp, BLK(i, 10), BLK(i, 11), h, h, BATCH * LTOK, EMBED, MLPD,
         MLPD, MLPD, EMBED, 1, 1, 0, 0, 0, 0, 0, 0, 1, ACT_NONE);
  }

  // -------- head --------
  k_ln<<<dim3(BATCH), 256, 0, stream>>>(h, (long long)LTOK * EMBED, normG, normB,
                                        1e-5f, clsln, EMBED, EMBED);
  gemm(stream, clsln, fcW, fcB, nullptr, fbuf, BATCH, 256, EMBED,
       EMBED, EMBED, 256, 1, 1, 0, 0, 0, 0, 0, 0, 1, ACT_NONE);
  k_bnscale<<<g1(BATCH * 256), 256, 0, stream>>>(fbuf, bnG, bnB, BATCH * 256);
  gemm(stream, fbuf, WG, nullptr, nullptr, (float*)d_out, BATCH, 32, 256,
       256, 256, 32, 1, 1, 0, 0, 0, 0, 0, 0, 1, ACT_TANH);
}
